// ShakespeareLSTM_23742579212620
// MI455X (gfx1250) — matrix operand order repaired and checked by reading, not compiled, not run on hardware
//
#include <hip/hip_runtime.h>
#include <hip/hip_bf16.h>

// ---------------------------------------------------------------------------
// AWD-LSTM forward for gfx1250 (MI455X), bf16 WMMA + fp32 accumulate.
//   B=16, S=256, E=300 (pad 320), H=1150 (pad 1152), 4H=4600 (pad 4608)
//   layer2: 4H=1200 (pad 1216)
// ---------------------------------------------------------------------------

typedef __attribute__((ext_vector_type(16))) __bf16       v16bf;
typedef __attribute__((ext_vector_type(8)))  float        v8f;
typedef __attribute__((ext_vector_type(8)))  unsigned short us8;

union FragBF {
    v16bf bf;
    us8   h[2];
};

__device__ __forceinline__ unsigned short f2bf(float f) {
    unsigned int u = __builtin_bit_cast(unsigned int, f);
    u += 0x7fffu + ((u >> 16) & 1u);          // round-to-nearest-even
    return (unsigned short)(u >> 16);
}

__device__ __forceinline__ float sigmoidf_(float x) {
    return 1.0f / (1.0f + __expf(-x));
}

// ---------------------------------------------------------------------------
// fp32 -> bf16 convert with zero padding: src[R,C] -> dst[Rp,Cp]
// ---------------------------------------------------------------------------
__global__ void cvt_pad_kernel(const float* __restrict__ src,
                               unsigned short* __restrict__ dst,
                               int R, int C, int Cp, int total) {
    int i = blockIdx.x * blockDim.x + threadIdx.x;
    if (i >= total) return;
    int r = i / Cp, c = i - r * Cp;
    float v = (r < R && c < C) ? src[(size_t)r * C + c] : 0.0f;
    dst[i] = f2bf(v);
}

// ---------------------------------------------------------------------------
// Embedding gather + sqrt(E) scale -> bf16 padded [4096, 320]
// ---------------------------------------------------------------------------
__global__ void embed_gather_kernel(const int* __restrict__ ids,
                                    const float* __restrict__ emb,
                                    unsigned short* __restrict__ x0) {
    int i = blockIdx.x * blockDim.x + threadIdx.x;   // 4096*320
    if (i >= 4096 * 320) return;
    int row = i / 320, c = i - row * 320;
    int id = ids[row];
    float v = (c < 300) ? emb[(size_t)id * 300 + c] * 17.320508075688775f : 0.0f;
    x0[i] = f2bf(v);
}

// ---------------------------------------------------------------------------
// GEMM: C[M,N] (fp32) = A[M,Kp](bf16,row-major) * W[N,Kp](bf16,row-major)^T
//       + optional bias0[n]+bias1[n].
// One wave per 16x64 output tile (1 M-tile x 4 N-tiles, shared A-frag).
// W buffers are padded to Ngroups*64 rows; stores are unguarded into padded C
// except bias reads, which are guarded by n<N.
// ---------------------------------------------------------------------------
__global__ void gemm_bf16_wmma(const unsigned short* __restrict__ A, int lda,
                               const unsigned short* __restrict__ W, int ldw,
                               float* __restrict__ C, int ldc,
                               const float* __restrict__ bias0,
                               const float* __restrict__ bias1,
                               int Mtiles, int Ngroups, int N, int Kp) {
    int wave = threadIdx.x >> 5;
    int lane = threadIdx.x & 31;
    int job  = blockIdx.x * (blockDim.x >> 5) + wave;
    if (job >= Mtiles * Ngroups) return;
    int ng = job / Mtiles;
    int mt = job - ng * Mtiles;
    int l16 = lane & 15, half = lane >> 4;

    const unsigned short* Ar = A + (size_t)(mt * 16 + l16) * lda + half * 16;
    const unsigned short* W0 = W + (size_t)(ng * 64 + l16) * ldw + half * 16;
    const unsigned short* W1 = W0 + (size_t)16 * ldw;
    const unsigned short* W2 = W1 + (size_t)16 * ldw;
    const unsigned short* W3 = W2 + (size_t)16 * ldw;

    v8f a0 = {}, a1 = {}, a2 = {}, a3 = {};
    for (int k = 0; k < Kp; k += 32) {
        __builtin_prefetch(W0 + k + 512, 0, 1);   // stream W through GL2
        FragBF fa, f0, f1, f2, f3;
        fa.h[0] = *(const us8*)(Ar + k);      fa.h[1] = *(const us8*)(Ar + k + 8);
        f0.h[0] = *(const us8*)(W0 + k);      f0.h[1] = *(const us8*)(W0 + k + 8);
        f1.h[0] = *(const us8*)(W1 + k);      f1.h[1] = *(const us8*)(W1 + k + 8);
        f2.h[0] = *(const us8*)(W2 + k);      f2.h[1] = *(const us8*)(W2 + k + 8);
        f3.h[0] = *(const us8*)(W3 + k);      f3.h[1] = *(const us8*)(W3 + k + 8);
        a0 = __builtin_amdgcn_wmma_f32_16x16x32_bf16(false, fa.bf, false, f0.bf, (short)0, a0, false, false);
        a1 = __builtin_amdgcn_wmma_f32_16x16x32_bf16(false, fa.bf, false, f1.bf, (short)0, a1, false, false);
        a2 = __builtin_amdgcn_wmma_f32_16x16x32_bf16(false, fa.bf, false, f2.bf, (short)0, a2, false, false);
        a3 = __builtin_amdgcn_wmma_f32_16x16x32_bf16(false, fa.bf, false, f3.bf, (short)0, a3, false, false);
    }

    int mBase = mt * 16 + half * 8;
    int nBase = ng * 64 + l16;
#pragma unroll
    for (int t = 0; t < 4; ++t) {
        int n = nBase + t * 16;
        float b = (bias0 && n < N) ? (bias0[n] + bias1[n]) : 0.0f;
        v8f acc = (t == 0) ? a0 : (t == 1) ? a1 : (t == 2) ? a2 : a3;
#pragma unroll
        for (int r = 0; r < 8; ++r)
            C[(size_t)(mBase + r) * ldc + n] = acc[r] + b;
    }
}

// ---------------------------------------------------------------------------
// One LSTM timestep. Each wave owns one 16-column h-tile and computes the
// four gate strips (i,f,g,o at cols g*H + tile) of  h_prev @ W_hh^T  via
// WMMA (M=16 batch rows == one tile), adds precomputed x_gates, applies the
// nonlinearities and updates c / h / layer output. Column-disjoint across
// waves, so no intra-step sync needed; h is ping-ponged across steps.
// W_hh (<=10.6 MB bf16) stays resident in the 192 MB L2 across all 256 steps.
// ---------------------------------------------------------------------------
__global__ void lstm_step_kernel(const unsigned short* __restrict__ hin,
                                 unsigned short* __restrict__ hout,
                                 const unsigned short* __restrict__ Whh, int ldw,
                                 const float* __restrict__ xg, int ldg,
                                 float* __restrict__ cst,
                                 unsigned short* __restrict__ xout, int ldo,
                                 int s, int H, int Kp, int nTiles) {
    int wave = threadIdx.x >> 5;
    int lane = threadIdx.x & 31;
    int t = blockIdx.x * (blockDim.x >> 5) + wave;
    if (t >= nTiles) return;
    int l16 = lane & 15, half = lane >> 4;

    const unsigned short* Ar = hin + (size_t)l16 * Kp + half * 16;
    const unsigned short* Wi = Whh + (size_t)(0 * H + t * 16 + l16) * ldw + half * 16;
    const unsigned short* Wf = Whh + (size_t)(1 * H + t * 16 + l16) * ldw + half * 16;
    const unsigned short* Wg = Whh + (size_t)(2 * H + t * 16 + l16) * ldw + half * 16;
    const unsigned short* Wo = Whh + (size_t)(3 * H + t * 16 + l16) * ldw + half * 16;

    v8f ai = {}, af = {}, ag = {}, ao = {};
    for (int k = 0; k < Kp; k += 32) {
        FragBF fa, fi, ff, fg, fo;
        fa.h[0] = *(const us8*)(Ar + k);  fa.h[1] = *(const us8*)(Ar + k + 8);
        fi.h[0] = *(const us8*)(Wi + k);  fi.h[1] = *(const us8*)(Wi + k + 8);
        ff.h[0] = *(const us8*)(Wf + k);  ff.h[1] = *(const us8*)(Wf + k + 8);
        fg.h[0] = *(const us8*)(Wg + k);  fg.h[1] = *(const us8*)(Wg + k + 8);
        fo.h[0] = *(const us8*)(Wo + k);  fo.h[1] = *(const us8*)(Wo + k + 8);
        ai = __builtin_amdgcn_wmma_f32_16x16x32_bf16(false, fa.bf, false, fi.bf, (short)0, ai, false, false);
        af = __builtin_amdgcn_wmma_f32_16x16x32_bf16(false, fa.bf, false, ff.bf, (short)0, af, false, false);
        ag = __builtin_amdgcn_wmma_f32_16x16x32_bf16(false, fa.bf, false, fg.bf, (short)0, ag, false, false);
        ao = __builtin_amdgcn_wmma_f32_16x16x32_bf16(false, fa.bf, false, fo.bf, (short)0, ao, false, false);
    }

    int nI = t * 16 + l16;         // i-gate column, also h column
#pragma unroll
    for (int r = 0; r < 8; ++r) {
        int b = half * 8 + r;                     // batch row (C/D layout)
        size_t row = (size_t)(b * 256 + s);       // (b,s) row of x_gates / output
        float xi = xg[row * ldg + nI];
        float xf = xg[row * ldg + nI + H];
        float xc = xg[row * ldg + nI + 2 * H];
        float xo = xg[row * ldg + nI + 3 * H];
        float gi = sigmoidf_(ai[r] + xi);
        float gf = sigmoidf_(af[r] + xf);
        float gc = tanhf(ag[r] + xc);
        float go = sigmoidf_(ao[r] + xo);
        if (nI < H) {
            float co = cst[(size_t)b * Kp + nI];
            float cn = gf * co + gi * gc;
            float hn = go * tanhf(cn);
            cst[(size_t)b * Kp + nI] = cn;
            unsigned short hb = f2bf(hn);
            hout[(size_t)b * Kp + nI] = hb;
            xout[row * ldo + nI] = hb;
        }
    }
}

// ---------------------------------------------------------------------------
// Host side
// ---------------------------------------------------------------------------
extern "C" void kernel_launch(void* const* d_in, const int* in_sizes, int n_in,
                              void* d_out, int out_size, void* d_ws, size_t ws_size,
                              hipStream_t stream) {
    (void)in_sizes; (void)n_in; (void)out_size; (void)ws_size;

    const int*   ids  = (const int*)d_in[0];
    const float* emb  = (const float*)d_in[1];
    const float* Wih[3] = {(const float*)d_in[2], (const float*)d_in[6],  (const float*)d_in[10]};
    const float* Whh[3] = {(const float*)d_in[3], (const float*)d_in[7],  (const float*)d_in[11]};
    const float* bih[3] = {(const float*)d_in[4], (const float*)d_in[8],  (const float*)d_in[12]};
    const float* bhh[3] = {(const float*)d_in[5], (const float*)d_in[9],  (const float*)d_in[13]};
    float* logits = (float*)d_out;

    constexpr int M = 4096;           // B*S
    constexpr int Mt = 256;           // M/16
    constexpr int E = 300, EP = 320;
    constexpr int H = 1150, HP = 1152;
    constexpr int G0 = 4600, G0P = 4608;   // 4H, padded to 64
    constexpr int G2 = 1200, G2P = 1216;
    constexpr int V = 32000;

    // ---- carve workspace -------------------------------------------------
    size_t off = 0;
    auto carve = [&](size_t bytes) {
        void* p = (char*)d_ws + off;
        off += (bytes + 255) & ~(size_t)255;
        return p;
    };
    unsigned short* embB  = (unsigned short*)carve((size_t)V   * EP * 2);
    unsigned short* Wih0b = (unsigned short*)carve((size_t)G0P * EP * 2);
    unsigned short* Whh0b = (unsigned short*)carve((size_t)G0P * HP * 2);
    unsigned short* Wih1b = (unsigned short*)carve((size_t)G0P * HP * 2);
    unsigned short* Whh1b = (unsigned short*)carve((size_t)G0P * HP * 2);
    unsigned short* Wih2b = (unsigned short*)carve((size_t)G2P * HP * 2);
    unsigned short* Whh2b = (unsigned short*)carve((size_t)G2P * EP * 2);
    unsigned short* x0    = (unsigned short*)carve((size_t)M * EP * 2);
    unsigned short* x1    = (unsigned short*)carve((size_t)M * HP * 2);
    unsigned short* x2    = (unsigned short*)carve((size_t)M * HP * 2);
    unsigned short* x3    = (unsigned short*)carve((size_t)M * EP * 2);
    float*          xg    = (float*)carve((size_t)M * G0P * 4);
    unsigned short* hA    = (unsigned short*)carve((size_t)16 * HP * 2);
    unsigned short* hB    = (unsigned short*)carve((size_t)16 * HP * 2);
    float*          cbuf  = (float*)carve((size_t)16 * HP * 4);

    auto cvt = [&](const float* src, unsigned short* dst, int R, int C, int Rp, int Cp) {
        int tot = Rp * Cp;
        cvt_pad_kernel<<<(tot + 255) / 256, 256, 0, stream>>>(src, dst, R, C, Cp, tot);
    };

    // ---- weight prep (fp32 -> bf16, zero-padded) -------------------------
    cvt(emb,    embB,  V,  E,    V,   EP);
    cvt(Wih[0], Wih0b, G0, E,    G0P, EP);
    cvt(Whh[0], Whh0b, G0, H,    G0P, HP);
    cvt(Wih[1], Wih1b, G0, H,    G0P, HP);
    cvt(Whh[1], Whh1b, G0, H,    G0P, HP);
    cvt(Wih[2], Wih2b, G2, H,    G2P, HP);
    cvt(Whh[2], Whh2b, G2, E,    G2P, EP);

    // ---- zero padded activation buffers (pad columns never written) ------
    hipMemsetAsync(x1, 0, (size_t)M * HP * 2, stream);
    hipMemsetAsync(x2, 0, (size_t)M * HP * 2, stream);
    hipMemsetAsync(x3, 0, (size_t)M * EP * 2, stream);

    // ---- embedding gather + scale ---------------------------------------
    embed_gather_kernel<<<(M * EP + 255) / 256, 256, 0, stream>>>(ids, emb, x0);

    struct LayerCfg {
        const unsigned short *xin, *wih, *whh;
        unsigned short *xout;
        int lda, kIn;              // input GEMM: lda / K
        int Npad, Ngroups, Nreal;  // gate dims
        int Hout, Kr;              // recurrent: H and padded H (ldw of Whh)
        int ldo, nTiles;
    };
    LayerCfg L[3] = {
        { x0, Wih0b, Whh0b, x1, EP, EP, G0P, G0P / 64, G0, H, HP, HP, HP / 16 },
        { x1, Wih1b, Whh1b, x2, HP, HP, G0P, G0P / 64, G0, H, HP, HP, HP / 16 },
        { x2, Wih2b, Whh2b, x3, HP, HP, G2P, G2P / 64, G2, E, EP, EP, EP / 16 },
    };

    for (int l = 0; l < 3; ++l) {
        const LayerCfg& cf = L[l];
        // x_gates = X @ W_ih^T + (b_ih + b_hh)
        int jobs = Mt * cf.Ngroups;
        gemm_bf16_wmma<<<(jobs + 7) / 8, 256, 0, stream>>>(
            cf.xin, cf.lda, cf.wih, cf.kIn, xg, cf.Npad,
            bih[l], bhh[l], Mt, cf.Ngroups, cf.Nreal, cf.kIn);

        // reset recurrent state
        hipMemsetAsync(hA,   0, (size_t)16 * HP * 2, stream);
        hipMemsetAsync(hB,   0, (size_t)16 * HP * 2, stream);
        hipMemsetAsync(cbuf, 0, (size_t)16 * HP * 4, stream);

        int blocks = (cf.nTiles + 3) / 4;   // 4 waves / block, 1 tile / wave
        for (int s = 0; s < 256; ++s) {
            unsigned short* hin  = (s & 1) ? hB : hA;
            unsigned short* hout = (s & 1) ? hA : hB;
            lstm_step_kernel<<<blocks, 128, 0, stream>>>(
                hin, hout, cf.whh, cf.Kr, xg, cf.Npad, cbuf,
                cf.xout, cf.ldo, s, cf.Hout, cf.Kr, cf.nTiles);
        }
    }

    // ---- tied-embedding projection: logits = x3 @ emb^T ------------------
    {
        int Ngroups = V / 64;               // 500
        int jobs = Mt * Ngroups;
        gemm_bf16_wmma<<<(jobs + 7) / 8, 256, 0, stream>>>(
            x3, EP, embB, EP, logits, V,
            nullptr, nullptr, Mt, Ngroups, V, EP);
    }
}
